// SpatialTemporalAtomDecoder_26697516712054
// MI455X (gfx1250) — matrix operand order repaired and checked by reading, not compiled, not run on hardware
//
#include <hip/hip_runtime.h>
#include <hip/hip_bf16.h>

// ---------------------------------------------------------------------------
// SpatialTemporalAtomDecoder for MI455X (gfx1250, wave32, WMMA).
// All heavy GEMMs use v_wmma_f32_16x16x32_f16 (f16 A/B staged in LDS, f32 acc).
// Memory-bound workload (~3GB f32 traffic vs ~0.4 TFLOP math) -> fuse epilogues,
// use b128 global loads for all staging, prefetch next K-chunk.
//
// Input order assumption (setup_inputs insertion order, params flattened):
//  0 a(16,512,768) 1 q(16,4096,128) 2 c(16,4096,128) 3 atom_dec_bias(2,128,32,128,12)
//  4 atom_to_token(2,4096,512) 5 atom_pad_mask(2,4096) 6 timestamps(2,8)
//  7 w_a2q 8 ad1_snorm_w 9 ad1_scale_w 10 ad1_scale_b 11 ad1_bias_w
// 12 wq 13 bq 14 wk 15 wv 16 wg 17 wo 18 op1_w 19 op1_b
// 20 ad2_snorm_w 21 ad2_scale_w 22 ad2_scale_b 23 ad2_bias_w
// 24 w_swish 25 w_a2b 26 w_b2a 27 op2_w 28 op2_b
// 29 t_ln_w 30 t_ln_b 31 t_qkv 32 t_decay 33 t_out
// 34 f_ln_w 35 f_ln_b 36 w_pos 37 T (unused; constants hardcoded)
// ---------------------------------------------------------------------------

typedef _Float16 f16;
typedef __attribute__((ext_vector_type(16))) _Float16 v16h;
typedef __attribute__((ext_vector_type(8)))  float    v8f;

#define DEV static __device__ __forceinline__

DEV float sigmoidf_(float x) { return 1.0f / (1.0f + __expf(-x)); }

DEV v8f wmma_f16(v16h a, v16h b, v8f c) {
  return __builtin_amdgcn_wmma_f32_16x16x32_f16(false, a, false, b, (short)0, c,
                                                false, false);
}

DEV void cvt4(f16* d, float4 v) {
  d[0] = (f16)v.x; d[1] = (f16)v.y; d[2] = (f16)v.z; d[3] = (f16)v.w;
}

// A-fragment K pattern for 16-bit 16x32 A (matches B):
//   k = 16*(lane>=16) + 2*p + sub
// B-fragment (K x 16, K=32): n = lane&15, k = 16*(lane>=16) + 2*p + sub
// C/D: n = lane&15, m = 8*(lane>=16) + vgpr

// ---------------------------------------------------------------------------
// Generic GEMM: C[r,n] = act(sum_k A[r,k]*W[k,n] + bias[n]); act 0=none 1=sigmoid
// REQUIRES R%32==0, K%32==0, N%64==0 (all call sites comply -> no guards).
// grid (R/32, N/64), block 256 (8 waves -> 2x4 16x16 tiles)
// ---------------------------------------------------------------------------
__global__ __launch_bounds__(256) void gemm_wmma_kernel(
    const float* __restrict__ A, const float* __restrict__ W,
    const float* __restrict__ bias, float* __restrict__ C,
    int R, int K, int N, int act) {
  __shared__ __align__(16) f16 As[32][36];
  __shared__ __align__(16) f16 Bt[64][36];
  int tid = threadIdx.x, lane = tid & 31, wave = tid >> 5;
  int rowBase = blockIdx.x * 32;
  int colBase = blockIdx.y * 64;
  int rowOff = (wave & 1) * 16;
  int colOff = (wave >> 1) * 16;
  int ar = tid >> 3, akq = (tid & 7) * 4;   // A: 4 floats/thread, row-wise
  int bn = tid & 63, bkb = (tid >> 6) * 8;  // W: 8 k-rows of one column/thread
  const float* aP = A + (long)(rowBase + ar) * K + akq;
  const float* wP = W + (long)bkb * N + colBase + bn;
  v8f acc = {};
  for (int kc = 0; kc < K; kc += 32) {
    if (kc + 32 < K) {
      __builtin_prefetch(aP + kc + 32, 0, 3);
      __builtin_prefetch(wP + (long)(kc + 32) * N, 0, 3);
    }
    __syncthreads();
    float4 av = *(const float4*)(aP + kc);
    cvt4(&As[ar][akq], av);
#pragma unroll
    for (int kk = 0; kk < 8; ++kk)
      Bt[bn][bkb + kk] = (f16)wP[(long)(kc + kk) * N];
    __syncthreads();
    v16h af, bf;
    int m = rowOff + (lane & 15);
    int khi = lane >> 4;
#pragma unroll
    for (int p = 0; p < 8; ++p) {
      int ka = (khi << 4) + (p << 1);
      af[2 * p] = As[m][ka];
      af[2 * p + 1] = As[m][ka + 1];
      int nb = colOff + (lane & 15);
      int kb = (khi << 4) + (p << 1);
      bf[2 * p] = Bt[nb][kb];
      bf[2 * p + 1] = Bt[nb][kb + 1];
    }
    acc = wmma_f16(af, bf, acc);
  }
  int cn = colBase + colOff + (lane & 15);
  long rb = rowBase + rowOff + ((lane >> 4) << 3);
  float bv = bias ? bias[cn] : 0.0f;
#pragma unroll
  for (int r8 = 0; r8 < 8; ++r8) {
    float v = acc[r8] + bv;
    if (act == 1) v = sigmoidf_(v);
    C[(rb + r8) * N + cn] = v;
  }
}

// ---------------------------------------------------------------------------
// AdaLN: Out = sigmoid(sn@sw + sb)*ln(X) + sn@bw, sn = ln(S)*snw.
// 16 rows/block, K=N=128 fixed. grid 65536/16=4096, block 256.
// ---------------------------------------------------------------------------
__global__ __launch_bounds__(256) void adaln_wmma_kernel(
    const float* __restrict__ X, const float* __restrict__ S,
    const float* __restrict__ snw, const float* __restrict__ sw,
    const float* __restrict__ sb, const float* __restrict__ bw,
    float* __restrict__ Out) {
  __shared__ float an[16][128];
  __shared__ __align__(16) f16 sn[16][132];
  __shared__ __align__(16) f16 B1[128][36];
  __shared__ __align__(16) f16 B2[128][36];
  __shared__ float redA[256], redB[256];
  int tid = threadIdx.x, lane = tid & 31, wave = tid >> 5;
  int team = tid >> 4, sub = tid & 15;
  long rowBase = (long)blockIdx.x * 16;
  long gx = (rowBase + team) * 128;
  float sx = 0, sxx = 0, ss = 0, sss = 0;
  for (int j = sub; j < 128; j += 16) {
    float vx = X[gx + j]; sx += vx; sxx += vx * vx;
    float vs = S[gx + j]; ss += vs; sss += vs * vs;
  }
  redA[tid] = sx; redB[tid] = sxx; __syncthreads();
  for (int st = 8; st; st >>= 1) {
    if (sub < st) { redA[tid] += redA[tid + st]; redB[tid] += redB[tid + st]; }
    __syncthreads();
  }
  float mux = redA[team << 4] * (1.f / 128.f);
  float rx = rsqrtf(redB[team << 4] * (1.f / 128.f) - mux * mux + 1e-5f);
  __syncthreads();
  redA[tid] = ss; redB[tid] = sss; __syncthreads();
  for (int st = 8; st; st >>= 1) {
    if (sub < st) { redA[tid] += redA[tid + st]; redB[tid] += redB[tid + st]; }
    __syncthreads();
  }
  float mus = redA[team << 4] * (1.f / 128.f);
  float rs = rsqrtf(redB[team << 4] * (1.f / 128.f) - mus * mus + 1e-5f);
  for (int j = sub; j < 128; j += 16) {
    an[team][j] = (X[gx + j] - mux) * rx;
    sn[team][j] = (f16)(((S[gx + j] - mus) * rs) * snw[j]);
  }
  int colOff = wave * 16;                    // 8 waves x 16 cols = 128
  int wn = tid & 127, wkb = (tid >> 7) * 16; // weight staging: 16 k-rows/col
  v8f acc1 = {}, acc2 = {};
  for (int kc = 0; kc < 128; kc += 32) {
    __syncthreads();
#pragma unroll
    for (int kk = 0; kk < 16; ++kk) {
      int gk = kc + wkb + kk;
      B1[wn][wkb + kk] = (f16)sw[gk * 128 + wn];
      B2[wn][wkb + kk] = (f16)bw[gk * 128 + wn];
    }
    __syncthreads();
    v16h af, b1f, b2f;
    int m = lane & 15, khi = lane >> 4;
#pragma unroll
    for (int p = 0; p < 8; ++p) {
      int ka = kc + (khi << 4) + (p << 1);
      af[2 * p] = sn[m][ka];
      af[2 * p + 1] = sn[m][ka + 1];
      int nb = colOff + (lane & 15);
      int kb = (khi << 4) + (p << 1);
      b1f[2 * p] = B1[nb][kb]; b1f[2 * p + 1] = B1[nb][kb + 1];
      b2f[2 * p] = B2[nb][kb]; b2f[2 * p + 1] = B2[nb][kb + 1];
    }
    acc1 = wmma_f16(af, b1f, acc1);
    acc2 = wmma_f16(af, b2f, acc2);
  }
  int cn = colOff + (lane & 15);
  int rb = (lane >> 4) << 3;
  float sbv = sb[cn];
#pragma unroll
  for (int r8 = 0; r8 < 8; ++r8) {
    int rl = rb + r8;
    float sig = sigmoidf_(acc1[r8] + sbv);
    Out[(rowBase + rl) * 128 + cn] = sig * an[rl][cn] + acc2[r8];
  }
}

// ---------------------------------------------------------------------------
// Fused SwiGLU: bt[r,n] = silu(t@Wsw[:,256+n]) * (t@Wsw[:,n]) * (t@Wab[:,n])
// K=128, N=256. grid (R/32, 4), block 256. Three WMMA chains share A-frags.
// ---------------------------------------------------------------------------
__global__ __launch_bounds__(256) void swiglu_wmma_kernel(
    const float* __restrict__ Tm, const float* __restrict__ Wsw,
    const float* __restrict__ Wab, float* __restrict__ BtOut) {
  __shared__ __align__(16) f16 As[32][36];
  __shared__ __align__(16) f16 B1[64][36];
  __shared__ __align__(16) f16 B2[64][36];
  __shared__ __align__(16) f16 B3[64][36];
  int tid = threadIdx.x, lane = tid & 31, wave = tid >> 5;
  long rowBase = (long)blockIdx.x * 32;
  int colBase = blockIdx.y * 64;
  int rowOff = (wave & 1) * 16;
  int colOff = (wave >> 1) * 16;
  int ar = tid >> 3, akq = (tid & 7) * 4;
  int bn = tid & 63, bkb = (tid >> 6) * 8;
  const float* aP = Tm + (rowBase + ar) * 128 + akq;
  const float* w1P = Wsw + (long)bkb * 512 + colBase + bn;
  const float* w3P = Wab + (long)bkb * 256 + colBase + bn;
  v8f a1 = {}, a2 = {}, a3 = {};
  for (int kc = 0; kc < 128; kc += 32) {
    __syncthreads();
    cvt4(&As[ar][akq], *(const float4*)(aP + kc));
#pragma unroll
    for (int kk = 0; kk < 8; ++kk) {
      B1[bn][bkb + kk] = (f16)w1P[(long)(kc + kk) * 512];
      B2[bn][bkb + kk] = (f16)w1P[(long)(kc + kk) * 512 + 256];
      B3[bn][bkb + kk] = (f16)w3P[(long)(kc + kk) * 256];
    }
    __syncthreads();
    v16h af, b1f, b2f, b3f;
    int m = rowOff + (lane & 15), khi = lane >> 4;
#pragma unroll
    for (int p = 0; p < 8; ++p) {
      int ka = (khi << 4) + (p << 1);
      af[2 * p] = As[m][ka]; af[2 * p + 1] = As[m][ka + 1];
      int nb = colOff + (lane & 15);
      int kb = (khi << 4) + (p << 1);
      b1f[2 * p] = B1[nb][kb]; b1f[2 * p + 1] = B1[nb][kb + 1];
      b2f[2 * p] = B2[nb][kb]; b2f[2 * p + 1] = B2[nb][kb + 1];
      b3f[2 * p] = B3[nb][kb]; b3f[2 * p + 1] = B3[nb][kb + 1];
    }
    a1 = wmma_f16(af, b1f, a1);
    a2 = wmma_f16(af, b2f, a2);
    a3 = wmma_f16(af, b3f, a3);
  }
  int cn = colBase + colOff + (lane & 15);
  long rb = rowBase + rowOff + ((lane >> 4) << 3);
#pragma unroll
  for (int r8 = 0; r8 < 8; ++r8) {
    float x1 = a1[r8], gt = a2[r8], ab = a3[r8];
    BtOut[(rb + r8) * 256 + cn] = (gt * sigmoidf_(gt)) * x1 * ab;
  }
}

// ---------------------------------------------------------------------------
// Spatial windowed attention: one block per (window wi in B*NW, head).
// QK^T and PV via WMMA; softmax + dec-bias + validity mask in between.
// Gate G (pre-sigmoided) applied at output. block 128 (4 waves).
// ---------------------------------------------------------------------------
__global__ __launch_bounds__(128) void spatial_attn_kernel(
    const float* __restrict__ Qb, const float* __restrict__ Kb,
    const float* __restrict__ Vb, const float* __restrict__ Gb,
    const float* __restrict__ zbias, const float* __restrict__ mask,
    float* __restrict__ Ob, int layer) {
  __shared__ __align__(16) f16 Qs[32][36];   // [query w][hd]
  __shared__ __align__(16) f16 Ks[128][36];  // [key j][hd]
  __shared__ __align__(16) f16 Vt[32][132];  // [hd][key j]
  __shared__ float Ssc[32][136];             // scores
  __shared__ __align__(16) f16 Ps[32][132];  // probs
  int tid = threadIdx.x, lane = tid & 31, wave = tid >> 5;
  int wi = blockIdx.x, head = blockIdx.y;
  int b = wi >> 7, nw = wi & 127;
  int b0 = b >> 3;
  long qrow0 = (long)wi * 32;
  int hoff = head * 32;
  {  // Q staging: 4 threads/row, 8 floats each (2x b128)
    int w = tid >> 2, seg = (tid & 3) * 8;
    const float* qp = Qb + (qrow0 + w) * 128 + hoff + seg;
    cvt4(&Qs[w][seg], *(const float4*)qp);
    cvt4(&Qs[w][seg + 4], *(const float4*)(qp + 4));
  }
  int center = nw * 32 + 16 - 64;  // nw*W + W/2 - H/2
  {  // K/V staging: one key row per thread, 8x b128 each
    int j = tid;
    int mi = center + j;
    int mc = mi < 0 ? 0 : (mi > 4095 ? 4095 : mi);
    const float* kp = Kb + ((long)b * 4096 + mc) * 128 + hoff;
    const float* vp = Vb + ((long)b * 4096 + mc) * 128 + hoff;
#pragma unroll
    for (int s = 0; s < 8; ++s) {
      cvt4(&Ks[j][s * 4], *(const float4*)(kp + s * 4));
      float4 vv = *(const float4*)(vp + s * 4);
      Vt[s * 4 + 0][j] = (f16)vv.x;
      Vt[s * 4 + 1][j] = (f16)vv.y;
      Vt[s * 4 + 2][j] = (f16)vv.z;
      Vt[s * 4 + 3][j] = (f16)vv.w;
    }
  }
  __syncthreads();
  const float scale = 0.17677669529663687f;  // 1/sqrt(32)
#pragma unroll
  for (int t = 0; t < 4; ++t) {  // 16 tiles: 2 row x 8 col, 4 per wave
    int tileId = wave * 4 + t;
    int rt = tileId & 1, ct = tileId >> 1;
    v16h af, bf;
    v8f acc = {};
    int m = rt * 16 + (lane & 15), khi = lane >> 4;
#pragma unroll
    for (int p = 0; p < 8; ++p) {
      int ka = (khi << 4) + (p << 1);
      af[2 * p] = Qs[m][ka]; af[2 * p + 1] = Qs[m][ka + 1];
      int nb = ct * 16 + (lane & 15);
      int kb = (khi << 4) + (p << 1);
      bf[2 * p] = Ks[nb][kb]; bf[2 * p + 1] = Ks[nb][kb + 1];
    }
    acc = wmma_f16(af, bf, acc);
    int cn = ct * 16 + (lane & 15);         // key index
    int rb = rt * 16 + ((lane >> 4) << 3);  // query base
    int mi = center + cn;
    int mc = mi < 0 ? 0 : (mi > 4095 ? 4095 : mi);
    float mval = (mi < 0 || mi > 4095) ? 0.0f : mask[b0 * 4096 + mc];
    float mb = (mval - 1.0f) * 1e9f;
#pragma unroll
    for (int r8 = 0; r8 < 8; ++r8) {
      int w = rb + r8;
      long zi = ((((long)b0 * 128 + nw) * 32 + w) * 128 + cn) * 12 + layer * 4 + head;
      Ssc[w][cn] = acc[r8] * scale + zbias[zi] + mb;
    }
  }
  __syncthreads();
  // softmax: 4 lanes per query row
  int row = tid >> 2, q4 = tid & 3;
  float mx = -3e38f;
  for (int j = q4; j < 128; j += 4) mx = fmaxf(mx, Ssc[row][j]);
  mx = fmaxf(mx, __shfl_xor(mx, 1, 32));
  mx = fmaxf(mx, __shfl_xor(mx, 2, 32));
  float sum = 0;
  for (int j = q4; j < 128; j += 4) {
    float e = __expf(Ssc[row][j] - mx);
    Ssc[row][j] = e; sum += e;
  }
  sum += __shfl_xor(sum, 1, 32);
  sum += __shfl_xor(sum, 2, 32);
  float inv = 1.0f / sum;
  for (int j = q4; j < 128; j += 4) Ps[row][j] = (f16)(Ssc[row][j] * inv);
  __syncthreads();
  {  // O = P(32x128) @ V(128x32): 2x2 tiles, one per wave
    int rt = wave & 1, ct = wave >> 1;
    v8f acc = {};
    for (int kc = 0; kc < 128; kc += 32) {
      v16h af, bf;
      int m = rt * 16 + (lane & 15), khi = lane >> 4;
#pragma unroll
      for (int p = 0; p < 8; ++p) {
        int ka = kc + (khi << 4) + (p << 1);
        af[2 * p] = Ps[m][ka]; af[2 * p + 1] = Ps[m][ka + 1];
        int nb = ct * 16 + (lane & 15);
        int kb = kc + (khi << 4) + (p << 1);
        bf[2 * p] = Vt[nb][kb]; bf[2 * p + 1] = Vt[nb][kb + 1];
      }
      acc = wmma_f16(af, bf, acc);
    }
    int cn = ct * 16 + (lane & 15);
    int rb = rt * 16 + ((lane >> 4) << 3);
#pragma unroll
    for (int r8 = 0; r8 < 8; ++r8) {
      long oi = (qrow0 + rb + r8) * 128 + hoff + cn;
      Ob[oi] = Gb[oi] * acc[r8];
    }
  }
}

// ---------------------------------------------------------------------------
// Temporal attention (T=8): one wave per (b0, m, head). VALU (tiny K).
// ---------------------------------------------------------------------------
__global__ __launch_bounds__(256) void temporal_attn_kernel(
    const float* __restrict__ qkv, const float* __restrict__ ts,
    const float* __restrict__ tdec, float* __restrict__ Ot) {
  __shared__ float qS[8][8][32];
  __shared__ float kS[8][8][32];
  __shared__ float vS[8][8][32];
  __shared__ float pS[8][8][8];
  int tid = threadIdx.x, lane = tid & 31, wave = tid >> 5;
  int unit = blockIdx.x * 8 + wave;  // b0*4096*4 + m*4 + head, 32768 total
  int head = unit & 3;
  int am = (unit >> 2) & 4095;
  int b0 = unit >> 14;
  const float scale = 0.17677669529663687f;
  {  // staging: lane -> (t, 8-elem segment), 2x b128 per tensor
    int t = lane >> 2, seg = (lane & 3) * 8;
    long row = ((long)(b0 * 8 + t) * 4096 + am) * 384 + head * 32 + seg;
    const float* qp = qkv + row;
    *(float4*)&qS[wave][t][seg] = *(const float4*)qp;
    *(float4*)&qS[wave][t][seg + 4] = *(const float4*)(qp + 4);
    *(float4*)&kS[wave][t][seg] = *(const float4*)(qp + 128);
    *(float4*)&kS[wave][t][seg + 4] = *(const float4*)(qp + 132);
    *(float4*)&vS[wave][t][seg] = *(const float4*)(qp + 256);
    *(float4*)&vS[wave][t][seg + 4] = *(const float4*)(qp + 260);
  }
  __syncthreads();
  float dec = tdec[head];
  float sp = dec > 20.0f ? dec : __logf(1.0f + __expf(dec));  // softplus
  float pr[2];
#pragma unroll
  for (int pi = 0; pi < 2; ++pi) {
    int pid = lane + pi * 32;
    int qt = pid >> 3, kt = pid & 7;
    float s = 0;
    for (int d = 0; d < 32; ++d) s += qS[wave][qt][d] * kS[wave][kt][d];
    float dt = fabsf(ts[b0 * 8 + qt] - ts[b0 * 8 + kt]);
    pr[pi] = s * scale - sp * dt;
  }
#pragma unroll
  for (int pi = 0; pi < 2; ++pi) {  // softmax over kt (8 consecutive lanes)
    float mx = pr[pi];
    mx = fmaxf(mx, __shfl_xor(mx, 1, 32));
    mx = fmaxf(mx, __shfl_xor(mx, 2, 32));
    mx = fmaxf(mx, __shfl_xor(mx, 4, 32));
    float e = __expf(pr[pi] - mx);
    float su = e;
    su += __shfl_xor(su, 1, 32);
    su += __shfl_xor(su, 2, 32);
    su += __shfl_xor(su, 4, 32);
    pr[pi] = e / su;
  }
  pS[wave][lane >> 3][lane & 7] = pr[0];
  pS[wave][4 + (lane >> 3)][lane & 7] = pr[1];
  __syncthreads();
  for (int i = lane; i < 256; i += 32) {
    int qt = i >> 5, d = i & 31;
    float o = 0;
#pragma unroll
    for (int kt = 0; kt < 8; ++kt) o += pS[wave][qt][kt] * vS[wave][kt][d];
    long orow = ((long)(b0 * 8 + qt) * 4096 + am) * 128;
    Ot[orow + head * 32 + d] = o;
  }
}

// ---------------------------------------------------------------------------
// Small helper kernels
// ---------------------------------------------------------------------------
__global__ void tokidx_kernel(const float* __restrict__ a2t,
                              int* __restrict__ tidx) {
  int i = blockIdx.x * blockDim.x + threadIdx.x;
  if (i < 8192) {
    const float* r = a2t + (long)i * 512;
    int t = 0;
    for (int j = 0; j < 512; ++j)
      if (r[j] > 0.5f) { t = j; break; }
    tidx[i] = t;
  }
}

// 4 rows per block of 128; each lane handles one float4.
__global__ __launch_bounds__(128) void gather_add_kernel(
    const float* __restrict__ q, const float* __restrict__ ap,
    const int* __restrict__ tidx, float* __restrict__ qs) {
  int row = blockIdx.x * 4 + (threadIdx.x >> 5);  // 0..65535
  int lane = threadIdx.x & 31;
  int b = row >> 12, m = row & 4095, b0 = b >> 3;
  int tok = tidx[b0 * 4096 + m];
  long o = (long)row * 128 + lane * 4;
  float4 qv = *(const float4*)(q + o);
  float4 av = *(const float4*)(ap + ((long)b * 512 + tok) * 128 + lane * 4);
  float4 r;
  r.x = qv.x + av.x; r.y = qv.y + av.y; r.z = qv.z + av.z; r.w = qv.w + av.w;
  *(float4*)(qs + o) = r;
}

// n must be a multiple of 4 (always 65536*128 here).
__global__ void gated_add_kernel(float* __restrict__ Q,
                                 const float* __restrict__ g,
                                 const float* __restrict__ X, long n4) {
  long i = (long)blockIdx.x * blockDim.x + threadIdx.x;
  long stride = (long)gridDim.x * blockDim.x;
  for (; i < n4; i += stride) {
    float4 qv = *(float4*)(Q + i * 4);
    float4 xv = *(const float4*)(X + i * 4);
    if (g) {
      float4 gv = *(const float4*)(g + i * 4);
      qv.x += gv.x * xv.x; qv.y += gv.y * xv.y;
      qv.z += gv.z * xv.z; qv.w += gv.w * xv.w;
    } else {
      qv.x += xv.x; qv.y += xv.y; qv.z += xv.z; qv.w += xv.w;
    }
    *(float4*)(Q + i * 4) = qv;
  }
}

__global__ __launch_bounds__(256) void ln_kernel(const float* __restrict__ X,
                                                 const float* __restrict__ w,
                                                 const float* __restrict__ b,
                                                 float* __restrict__ Y) {
  int row = blockIdx.x * 8 + (threadIdx.x >> 5);
  int lane = threadIdx.x & 31;
  long g = (long)row * 128;
  float s1 = 0, s2 = 0;
  for (int j = lane; j < 128; j += 32) { float v = X[g + j]; s1 += v; s2 += v * v; }
  for (int st = 16; st; st >>= 1) {
    s1 += __shfl_xor(s1, st, 32);
    s2 += __shfl_xor(s2, st, 32);
  }
  float mu = s1 * (1.f / 128.f);
  float r = rsqrtf(s2 * (1.f / 128.f) - mu * mu + 1e-5f);
  for (int j = lane; j < 128; j += 32) {
    float v = (X[g + j] - mu) * r;
    if (w) v *= w[j];
    if (b) v += b[j];
    Y[g + j] = v;
  }
}

__global__ __launch_bounds__(256) void final_head_kernel(
    const float* __restrict__ X, const float* __restrict__ w,
    const float* __restrict__ b, const float* __restrict__ wp,
    float* __restrict__ out) {
  int row = blockIdx.x * 8 + (threadIdx.x >> 5);
  int lane = threadIdx.x & 31;
  long g = (long)row * 128;
  float s1 = 0, s2 = 0;
  for (int j = lane; j < 128; j += 32) { float v = X[g + j]; s1 += v; s2 += v * v; }
  for (int st = 16; st; st >>= 1) {
    s1 += __shfl_xor(s1, st, 32);
    s2 += __shfl_xor(s2, st, 32);
  }
  float mu = s1 * (1.f / 128.f);
  float r = rsqrtf(s2 * (1.f / 128.f) - mu * mu + 1e-5f);
  float d0 = 0, d1 = 0, d2 = 0;
  for (int j = lane; j < 128; j += 32) {
    float v = (X[g + j] - mu) * r * w[j] + b[j];
    d0 += v * wp[j * 3 + 0];
    d1 += v * wp[j * 3 + 1];
    d2 += v * wp[j * 3 + 2];
  }
  for (int st = 16; st; st >>= 1) {
    d0 += __shfl_xor(d0, st, 32);
    d1 += __shfl_xor(d1, st, 32);
    d2 += __shfl_xor(d2, st, 32);
  }
  if (lane == 0) {
    out[(long)row * 3 + 0] = d0;
    out[(long)row * 3 + 1] = d1;
    out[(long)row * 3 + 2] = d2;
  }
}

// ---------------------------------------------------------------------------
// Host orchestration
// ---------------------------------------------------------------------------
static inline void gemm(const float* A, const float* W, const float* bias,
                        float* C, int R, int K, int N, int act,
                        hipStream_t s) {
  dim3 g(R / 32, N / 64);
  gemm_wmma_kernel<<<g, 256, 0, s>>>(A, W, bias, C, R, K, N, act);
}

extern "C" void kernel_launch(void* const* d_in, const int* in_sizes, int n_in,
                              void* d_out, int out_size, void* d_ws,
                              size_t ws_size, hipStream_t stream) {
  (void)in_sizes; (void)n_in; (void)out_size; (void)ws_size;
  constexpr long RM = 65536;  // B*M rows
  const float* a = (const float*)d_in[0];
  const float* q = (const float*)d_in[1];
  const float* c = (const float*)d_in[2];
  const float* adb = (const float*)d_in[3];
  const float* a2t = (const float*)d_in[4];
  const float* mask = (const float*)d_in[5];
  const float* ts = (const float*)d_in[6];
  const float* w_a2q = (const float*)d_in[7];
  const float* ad1_snw = (const float*)d_in[8];
  const float* ad1_sw = (const float*)d_in[9];
  const float* ad1_sb = (const float*)d_in[10];
  const float* ad1_bw = (const float*)d_in[11];
  const float* wq = (const float*)d_in[12];
  const float* bq = (const float*)d_in[13];
  const float* wk = (const float*)d_in[14];
  const float* wv = (const float*)d_in[15];
  const float* wg = (const float*)d_in[16];
  const float* wo = (const float*)d_in[17];
  const float* op1w = (const float*)d_in[18];
  const float* op1b = (const float*)d_in[19];
  const float* ad2_snw = (const float*)d_in[20];
  const float* ad2_sw = (const float*)d_in[21];
  const float* ad2_sb = (const float*)d_in[22];
  const float* ad2_bw = (const float*)d_in[23];
  const float* wsw = (const float*)d_in[24];
  const float* wa2b = (const float*)d_in[25];
  const float* wb2a = (const float*)d_in[26];
  const float* op2w = (const float*)d_in[27];
  const float* op2b = (const float*)d_in[28];
  const float* tlnw = (const float*)d_in[29];
  const float* tlnb = (const float*)d_in[30];
  const float* tqkv = (const float*)d_in[31];
  const float* tdec = (const float*)d_in[32];
  const float* tout = (const float*)d_in[33];
  const float* flnw = (const float*)d_in[34];
  const float* flnb = (const float*)d_in[35];
  const float* wpos = (const float*)d_in[36];
  float* out = (float*)d_out;

  // Workspace layout (~340MB), every region written before read each call.
  float* ws = (float*)d_ws;
  float* ap = ws;                        // 8192*128
  int* tidx = (int*)(ap + 8192L * 128);  // 8192
  float* qstate = (float*)(tidx + 8192); // RM*128
  float* bmat = qstate + RM * 128;       // RM*128
  float* qbuf = bmat + RM * 128;         // RM*128 } contiguous 96MB
  float* kbuf = qbuf + RM * 128;         // RM*128 }  reused as qkvbuf
  float* vbuf = kbuf + RM * 128;         // RM*128 }
  float* gbuf = vbuf + RM * 128;         // RM*128
  float* obuf = gbuf + RM * 128;         // RM*128
  float* xbuf = obuf + RM * 128;         // RM*128
  float* btbuf = xbuf + RM * 128;        // RM*256

  // token projection + gather into q
  gemm(a, w_a2q, nullptr, ap, 8192, 768, 128, 0, stream);
  tokidx_kernel<<<32, 256, 0, stream>>>(a2t, tidx);
  gather_add_kernel<<<16384, 128, 0, stream>>>(q, ap, tidx, qstate);

  for (int i = 0; i < 3; ++i) {
    long dd = (long)i * 128 * 128;
    // spatial: AdaLN -> Q/K/V/G -> windowed attention -> out proj + gated res
    adaln_wmma_kernel<<<4096, 256, 0, stream>>>(qstate, c, ad1_snw + i * 128,
        ad1_sw + dd, ad1_sb + i * 128, ad1_bw + dd, bmat);
    gemm(bmat, wq + dd, bq + i * 128, qbuf, 65536, 128, 128, 0, stream);
    gemm(bmat, wk + dd, nullptr, kbuf, 65536, 128, 128, 0, stream);
    gemm(bmat, wv + dd, nullptr, vbuf, 65536, 128, 128, 0, stream);
    gemm(bmat, wg + dd, nullptr, gbuf, 65536, 128, 128, 1, stream);
    spatial_attn_kernel<<<dim3(2048, 4), 128, 0, stream>>>(qbuf, kbuf, vbuf,
        gbuf, adb, mask, obuf, i);
    gemm(obuf, wo + dd, nullptr, xbuf, 65536, 128, 128, 0, stream);
    gemm(c, op1w + dd, op1b + i * 128, gbuf, 65536, 128, 128, 1, stream);
    gated_add_kernel<<<4096, 256, 0, stream>>>(qstate, gbuf, xbuf,
                                               RM * 128 / 4);
    // conditioned SwiGLU transition
    adaln_wmma_kernel<<<4096, 256, 0, stream>>>(qstate, c, ad2_snw + i * 128,
        ad2_sw + dd, ad2_sb + i * 128, ad2_bw + dd, bmat);
    swiglu_wmma_kernel<<<dim3(2048, 4), 256, 0, stream>>>(bmat,
        wsw + (long)i * 128 * 512, wa2b + (long)i * 128 * 256, btbuf);
    gemm(btbuf, wb2a + (long)i * 256 * 128, nullptr, xbuf, 65536, 256, 128, 0,
         stream);
    gemm(c, op2w + dd, op2b + i * 128, gbuf, 65536, 128, 128, 1, stream);
    gated_add_kernel<<<4096, 256, 0, stream>>>(qstate, gbuf, xbuf,
                                               RM * 128 / 4);
    // temporal attention with per-head decay
    ln_kernel<<<8192, 256, 0, stream>>>(qstate, tlnw + i * 128, tlnb + i * 128,
                                        bmat);
    gemm(bmat, tqkv + (long)i * 128 * 384, nullptr, qbuf /*qkv*/, 65536, 128,
         384, 0, stream);
    temporal_attn_kernel<<<4096, 256, 0, stream>>>(qbuf, ts, tdec + i * 4,
                                                   obuf);
    gemm(obuf, tout + dd, nullptr, xbuf, 65536, 128, 128, 0, stream);
    gated_add_kernel<<<4096, 256, 0, stream>>>(qstate, nullptr, xbuf,
                                               RM * 128 / 4);
  }
  final_head_kernel<<<8192, 256, 0, stream>>>(qstate, flnw, flnb, wpos, out);
}
